// NMO_V3_19009525252635
// MI455X (gfx1250) — compile-verified, matrix operand order checked by reading, not run on hardware
//
#include <hip/hip_runtime.h>

// ---------------------------------------------------------------------------
// Problem constants (from reference): B=1, T=48, N=512, D=128, H=4, hd=32,
// F=2048, L=4.  M_TOT = B*T*N = 24576 rows of the activation matrix.
// ---------------------------------------------------------------------------
#define M_TOT   24576
#define SEQ_N   512
#define DMODEL  128
#define NHEAD   4
#define HDIM    32
#define FFDIM   2048
#define NLAYER  4

typedef unsigned short u16;
typedef unsigned int   u32;
typedef unsigned long long u64;
typedef __attribute__((ext_vector_type(16))) __bf16 v16bf;
typedef __attribute__((ext_vector_type(8)))  float  v8f;
typedef __attribute__((ext_vector_type(4)))  unsigned int v4u;
typedef __attribute__((ext_vector_type(4)))  int v4i;
typedef __attribute__((ext_vector_type(8)))  int v8i_t;

// Tensor Data Mover availability (device pass only; manual-copy fallback)
#if defined(__HIP_DEVICE_COMPILE__) && __has_builtin(__builtin_amdgcn_tensor_load_to_lds)
#define HAVE_TDM 1
#else
#define HAVE_TDM 0
#endif

// ---------------------------------------------------------------------------
// bf16 helpers (round-to-nearest-even)
// ---------------------------------------------------------------------------
__device__ __forceinline__ u16 f2bf(float f) {
    u32 u = __float_as_uint(f);
    u32 r = u + 0x7FFFu + ((u >> 16) & 1u);
    return (u16)(r >> 16);
}
__device__ __forceinline__ u32 f2bf_pk(float lo, float hi) {
    return (u32)f2bf(lo) | ((u32)f2bf(hi) << 16);
}
__device__ __forceinline__ float bf2f(u16 v) {
    return __uint_as_float(((u32)v) << 16);
}
__device__ __forceinline__ float gelu_exact(float x) {
    return 0.5f * x * (1.0f + erff(x * 0.70710678118654752f));
}
__device__ __forceinline__ v8f vzero8() {
    v8f z = {0.f, 0.f, 0.f, 0.f, 0.f, 0.f, 0.f, 0.f};
    return z;
}
__device__ __forceinline__ v8f wmma_bf16(v16bf a, v16bf b, v8f c) {
    // D(16x16,f32) = A(16x32,bf16) x B(32x16,bf16) + C
    return __builtin_amdgcn_wmma_f32_16x16x32_bf16(
        false, a, false, b, (short)0, c, false, false);
}

// ---------------------------------------------------------------------------
// TDM: 2D tile load (global -> LDS) with per-row LDS padding.
// D# built per CDNA5 ISA ch.8: Group0 = {count, lds_addr, global_addr, type},
// Group1 = {flags/pad, tensor dims, tile dims, dim0 stride}.
//   pad_interval_code: dwords-per-interval = 2 << code
//   pad_amount_code:   pad dwords = code + 1
// Issue from one wave only; sync with s_wait_tensorcnt 0 + barrier.
// ---------------------------------------------------------------------------
__device__ __forceinline__ u32 lds_off(const void* p) {
    // generic (flat) LDS address: low 32 bits are the wave-relative LDS offset
    return (u32)(size_t)p;
}

__device__ __forceinline__ void tdm_load_2d(u32 lds_addr, const void* gptr,
                                            u32 tile_cols, u32 tile_rows,
                                            u32 row_stride_elems,
                                            u32 pad_interval_code,
                                            u32 pad_amount_code) {
#if HAVE_TDM
    u64 ga = (u64)(size_t)gptr;
    v4u g0;
    g0.x = 1u;                                  // count=1 (valid user descriptor)
    g0.y = lds_addr;                            // LDS byte address
    g0.z = (u32)ga;                             // global_addr[31:0]
    g0.w = (u32)(ga >> 32) | (2u << 30);        // global_addr[56:32] | type=2
    const u32 td0 = 0x40000000u, td1 = 0x40000000u;   // huge tensor dims (no clip)
    v8i_t g1;
    g1[0] = (int)((1u << 16) |                  // data_size = 1 -> 2 bytes
                  (1u << 20) |                  // pad_enable
                  (pad_interval_code << 22) | (pad_amount_code << 25));
    g1[1] = (int)((td0 & 0xFFFFu) << 16);                           // tensor_dim0 lo
    g1[2] = (int)(((td0 >> 16) & 0xFFFFu) | ((td1 & 0xFFFFu) << 16));
    g1[3] = (int)(((td1 >> 16) & 0xFFFFu) | ((tile_cols & 0xFFFFu) << 16)); // tile_dim0
    g1[4] = (int)(tile_rows & 0xFFFFu);                             // tile_dim1 (dim2=0)
    g1[5] = (int)row_stride_elems;                                  // dim0_stride[31:0]
    g1[6] = 0;
    g1[7] = 0;
    v4i gz = {0, 0, 0, 0};
#if defined(__clang_major__) && __clang_major__ >= 23
    v8i_t gz8 = {0, 0, 0, 0, 0, 0, 0, 0};
    __builtin_amdgcn_tensor_load_to_lds(g0, g1, gz, gz, gz8, 0);
#else
    __builtin_amdgcn_tensor_load_to_lds(g0, g1, gz, gz, 0);
#endif
#else
    (void)lds_addr; (void)gptr; (void)tile_cols; (void)tile_rows;
    (void)row_stride_elems; (void)pad_interval_code; (void)pad_amount_code;
#endif
}

__device__ __forceinline__ void tdm_wait0() {
#if HAVE_TDM
#if __has_builtin(__builtin_amdgcn_s_wait_tensorcnt)
    __builtin_amdgcn_s_wait_tensorcnt(0);
#endif
#endif
}

// ---------------------------------------------------------------------------
// WMMA fragment loaders per CDNA5 ISA 7.12.2 (wave32):
// A (16x32 bf16): lanes l and l+16 hold row M=l; VGPR p holds K pair
//   p<4 -> K = hi*8 + 2p, else K = 16 + hi*8 + 2(p-4).
// B (32x16 bf16): mirrors A with lane = column N.
// C (16x16 f32): VGPR r -> (M = r + 8*hi, N = lane&15).
// ---------------------------------------------------------------------------
__device__ __forceinline__ int frag_k(int p, int hi) {
    return (p < 4) ? (hi * 8 + 2 * p) : (16 + hi * 8 + 2 * (p - 4));
}

__device__ __forceinline__ v16bf frag_a_lds(const u16* base, int stride, int lane) {
    int l = lane & 15, hi = lane >> 4;
    union { v16bf v; u32 d[8]; } f;
    const u16* rp = base + l * stride;       // stride even -> dword aligned
#pragma unroll
    for (int p = 0; p < 8; ++p) {
        int k = frag_k(p, hi);               // even
        f.d[p] = *(const u32*)(rp + k);      // packed K pair
    }
    return f.v;
}

__device__ __forceinline__ v16bf frag_b_lds(const u16* base, int stride, int ncol, int lane) {
    int l = lane & 15, hi = lane >> 4;
    union { v16bf v; u16 u[16]; } f;
#pragma unroll
    for (int p = 0; p < 8; ++p) {
        int k = frag_k(p, hi);
        f.u[2 * p]     = base[(size_t)k       * stride + ncol + l];
        f.u[2 * p + 1] = base[(size_t)(k + 1) * stride + ncol + l];
    }
    return f.v;
}

// ---------------------------------------------------------------------------
// fp32 -> bf16 weight conversion
// ---------------------------------------------------------------------------
__global__ void k_cvt(const float* __restrict__ s, u16* __restrict__ d, int n) {
    int i = blockIdx.x * blockDim.x + threadIdx.x;
    if (i < n) d[i] = f2bf(s[i]);
}

// ---------------------------------------------------------------------------
// Embedding: feat = x @ node_proj_w + node_proj_b + time_emb[:, t]
// ---------------------------------------------------------------------------
__global__ void k_embed(const float* __restrict__ x, const float* __restrict__ W,
                        const float* __restrict__ b, const float* __restrict__ temb,
                        float* __restrict__ f32, u16* __restrict__ fb16) {
    int idx = blockIdx.x * blockDim.x + threadIdx.x;
    if (idx >= M_TOT * DMODEL) return;
    int row = idx >> 7;
    int d   = idx & 127;
    int t   = row >> 9;
    const float* xr = x + (size_t)row * 6;
    float a = b[d] + temb[(size_t)t * DMODEL + d];
#pragma unroll
    for (int c = 0; c < 6; ++c) a += xr[c] * W[c * DMODEL + d];
    f32[idx]  = a;
    fb16[idx] = f2bf(a);
}

// ---------------------------------------------------------------------------
// LayerNorm over D=128 (one row per 128-thread block), bf16 output
// ---------------------------------------------------------------------------
__global__ void k_ln(const float* __restrict__ f, const float* __restrict__ g,
                     const float* __restrict__ b, u16* __restrict__ y) {
    __shared__ float sd[128];
    int row = blockIdx.x, t = threadIdx.x;
    float v = f[(size_t)row * DMODEL + t];
    sd[t] = v;
    __syncthreads();
    for (int s = 64; s > 0; s >>= 1) { if (t < s) sd[t] += sd[t + s]; __syncthreads(); }
    float mu = sd[0] * (1.0f / 128.0f);
    __syncthreads();
    float d = v - mu;
    sd[t] = d * d;
    __syncthreads();
    for (int s = 64; s > 0; s >>= 1) { if (t < s) sd[t] += sd[t + s]; __syncthreads(); }
    float var = sd[0] * (1.0f / 128.0f);
    float r = rsqrtf(var + 1e-5f);
    y[(size_t)row * DMODEL + t] = f2bf(d * r * g[t] + b[t]);
}

// ---------------------------------------------------------------------------
// Generic bf16 WMMA GEMM:  C(M x N) = A(M x K) @ B(K x N)
//   tile: 128(M) x 64(N) per 256-thread workgroup, K chunked by 32,
//   TDM-staged + double-buffered when available.
// AMODE: 0 plain row-major A (TDM); 1 "exchange" gather (k<128 -> roll(feat,1)).
// EPI:   0 bias -> bf16; 1 gelu(bias [+ u*wu]) -> bf16;
//        2 residual: Cf32 += acc+bias, refresh bf16 mirror.
// ---------------------------------------------------------------------------
#define GM 128
#define GN 64
#define GK 32
#define ASTR 34   // 17 dwords/row: conflict-free; TDM pad(interval=16dw,+1dw)
#define BSTR 66   // 33 dwords/row: conflict-free; TDM pad(interval=32dw,+1dw)

template <int AMODE>
__device__ __forceinline__ void stage_a(u16* dst, const u16* __restrict__ A,
                                        int lda, int m0, int kt, int tid) {
    if (AMODE == 0) {
#if HAVE_TDM
        if ((tid >> 5) == 0)
            tdm_load_2d(lds_off(dst), A + (size_t)m0 * lda + kt,
                        GK, GM, (u32)lda, 3 /*16 dw*/, 0 /*+1 dw*/);
        return;
#endif
    }
    // manual path (always used for AMODE==1; fallback for AMODE==0)
    int r = tid >> 1, half = tid & 1;
    int grow = m0 + r;
    int gk   = kt + half * 16;
    const u16* src;
    if (AMODE == 1) {
        int seq = grow >> 9, n = grow & (SEQ_N - 1);
        int srow = (gk < 128) ? ((seq << 9) + ((n + SEQ_N - 1) & (SEQ_N - 1))) : grow;
        int scol = (gk < 128) ? gk : (gk - 128);
        src = A + (size_t)srow * lda + scol;
    } else {
        src = A + (size_t)grow * lda + gk;
    }
    const uint4* s4 = (const uint4*)src;
    uint4 p0 = s4[0], p1 = s4[1];
    u32* d = (u32*)&dst[r * ASTR + half * 16];
    d[0] = p0.x; d[1] = p0.y; d[2] = p0.z; d[3] = p0.w;
    d[4] = p1.x; d[5] = p1.y; d[6] = p1.z; d[7] = p1.w;
}

__device__ __forceinline__ void stage_b(u16* dst, const u16* __restrict__ Bw,
                                        int ldb, int n0, int kt, int tid) {
#if HAVE_TDM
    if ((tid >> 5) == 0)
        tdm_load_2d(lds_off(dst), Bw + (size_t)kt * ldb + n0,
                    GN, GK, (u32)ldb, 4 /*32 dw*/, 0 /*+1 dw*/);
#else
    int r = tid >> 3, c8 = (tid & 7) * 8;
    const uint4* s4 = (const uint4*)(Bw + (size_t)(kt + r) * ldb + n0 + c8);
    uint4 p = s4[0];
    u32* d = (u32*)&dst[r * BSTR + c8];
    d[0] = p.x; d[1] = p.y; d[2] = p.z; d[3] = p.w;
#endif
}

template <int AMODE, int EPI, bool HASU>
__global__ __launch_bounds__(256)
void k_gemm(const u16* __restrict__ A, int lda,
            const u16* __restrict__ Bw, int ldb, int K,
            const float* __restrict__ bias,
            float* __restrict__ Cf32,
            u16* __restrict__ Co, int ldo,
            const float* __restrict__ xin,   // x (for u_now) when HASU
            const float* __restrict__ wu) {  // ex_w1 row 256 when HASU
    __shared__ u16 As[2][GM * ASTR];
    __shared__ u16 Bs[2][GK * BSTR];

    int tid  = threadIdx.x;
    int wave = tid >> 5, lane = tid & 31;
    int m0 = blockIdx.x * GM, n0 = blockIdx.y * GN;

    v8f acc[4];
#pragma unroll
    for (int i = 0; i < 4; ++i) acc[i] = vzero8();

    const int nk = K / GK;

    // prologue: stage chunk 0 into buffer 0
    stage_a<AMODE>(As[0], A, lda, m0, 0, tid);
    stage_b(Bs[0], Bw, ldb, n0, 0, tid);

    for (int i = 0; i < nk; ++i) {
        int cur = i & 1;
        if (wave == 0) tdm_wait0();          // chunk i staged (TDM path)
        __syncthreads();                     // visible to all waves
        if (i + 1 < nk) {                    // overlap: stage chunk i+1
            stage_a<AMODE>(As[cur ^ 1], A, lda, m0, (i + 1) * GK, tid);
            stage_b(Bs[cur ^ 1], Bw, ldb, n0, (i + 1) * GK, tid);
        }
        v16bf afrag = frag_a_lds(As[cur] + wave * 16 * ASTR, ASTR, lane);
#pragma unroll
        for (int n = 0; n < 4; ++n) {
            v16bf bfrag = frag_b_lds(Bs[cur], BSTR, n * 16, lane);
            acc[n] = wmma_bf16(afrag, bfrag, acc[n]);
        }
        __syncthreads();                     // done reading buf[cur]
    }

    // ---- epilogue (C layout: VGPR r -> row r + 8*hi, col lane&15)
    int l = lane & 15, hi = lane >> 4;
    int rowb = m0 + wave * 16;
#pragma unroll
    for (int n = 0; n < 4; ++n) {
        int col = n0 + n * 16 + l;
        float bv = bias[col];
#pragma unroll
        for (int r = 0; r < 8; ++r) {
            int row = rowb + r + hi * 8;
            float v = acc[n][r] + bv;
            if (HASU) v += xin[(size_t)row * 6 + 2] * wu[col];
            if (EPI == 1) v = gelu_exact(v);
            if (EPI == 2) {
                float fv = Cf32[(size_t)row * DMODEL + col] + v;
                Cf32[(size_t)row * DMODEL + col] = fv;
                Co[(size_t)row * ldo + col] = f2bf(fv);
            } else {
                Co[(size_t)row * ldo + col] = f2bf(v);
            }
        }
    }
}

// ---------------------------------------------------------------------------
// Flash attention per (sequence, head): N=512 keys, hd=32, causal.
// K tile TDM-staged into LDS; Q/V fragments gathered from global (L2).
// Each of 8 waves owns one 16-row query tile per pass (4 passes).
// ---------------------------------------------------------------------------
#define KSTR 34
#define SSTR 33

__global__ __launch_bounds__(256)
void k_attn(const u16* __restrict__ qkv, u16* __restrict__ o) {
    __shared__ u16   Ks[SEQ_N * KSTR];
    __shared__ float Sw[8][16 * SSTR];   // per-wave 16x32 score/prob tile
    __shared__ float St[8][48];          // [0:16)=rowmax [16:32)=rowsum [32:48)=fac

    int seq = blockIdx.x, h = blockIdx.y;
    int tid = threadIdx.x, wave = tid >> 5, lane = tid & 31;
    int l = lane & 15, hi = lane >> 4;

    const u16* qbase = qkv + (size_t)seq * SEQ_N * 384 + h * HDIM;
    const u16* kbase = qbase + 128;
    const u16* vbase = qbase + 256;

    // ---- stage K (512 x 32 bf16, padded rows) into LDS
#if HAVE_TDM
    if (wave == 0) {
        tdm_load_2d(lds_off(Ks), kbase, HDIM, SEQ_N, 384, 3, 0);
        tdm_wait0();
    }
#else
    for (int kk = tid; kk < SEQ_N; kk += 256) {
        const uint4* src = (const uint4*)(kbase + (size_t)kk * 384);
        uint4 a = src[0], b = src[1], c = src[2], d = src[3];
        u32* dst = (u32*)&Ks[kk * KSTR];
        dst[0] = a.x;  dst[1] = a.y;  dst[2] = a.z;  dst[3] = a.w;
        dst[4] = b.x;  dst[5] = b.y;  dst[6] = b.z;  dst[7] = b.w;
        dst[8] = c.x;  dst[9] = c.y;  dst[10] = c.z; dst[11] = c.w;
        dst[12] = d.x; dst[13] = d.y; dst[14] = d.z; dst[15] = d.w;
    }
#endif
    __syncthreads();

    const float scale = 0.17677669529663687f;   // 1/sqrt(32)

    for (int pass = 0; pass < 4; ++pass) {
        int qt = pass * 8 + wave;      // query tile 0..31
        int qrow0 = qt * 16;

        // Q fragment (A-layout) gathered from global, packed-pair loads
        union { v16bf v; u32 d[8]; } qa;
        {
            const u16* qp = qbase + (size_t)(qrow0 + l) * 384;
#pragma unroll
            for (int p = 0; p < 8; ++p)
                qa.d[p] = *(const u32*)(qp + frag_k(p, hi));
        }

        v8f oacc0 = vzero8(), oacc1 = vzero8();
        if (lane < 16) { St[wave][lane] = -1e30f; St[wave][16 + lane] = 0.f; }
        __builtin_amdgcn_wave_barrier();

        int nkb = (qt >> 1) + 1;       // causal: key blocks of 32
        for (int kb = 0; kb < nkb; ++kb) {
            int key0 = kb * 32;

            v16bf kf0 = frag_a_lds(Ks + (size_t)key0 * KSTR,        KSTR, lane);
            v16bf kf1 = frag_a_lds(Ks + (size_t)(key0 + 16) * KSTR, KSTR, lane);
            v8f s0 = wmma_bf16(qa.v, kf0, vzero8());
            v8f s1 = wmma_bf16(qa.v, kf1, vzero8());

            // scatter scores to per-wave LDS tile
            float* sw = Sw[wave];
#pragma unroll
            for (int r = 0; r < 8; ++r) {
                sw[(r + hi * 8) * SSTR + l]      = s0[r];
                sw[(r + hi * 8) * SSTR + 16 + l] = s1[r];
            }
            __builtin_amdgcn_wave_barrier();

            // online softmax: lanes 0..15 each own one query row
            if (lane < 16) {
                int qrow = qrow0 + lane;
                float mo = St[wave][lane];
                float* rp = sw + lane * SSTR;
                float mx = mo;
#pragma unroll
                for (int c = 0; c < 32; ++c) {
                    float s = rp[c] * scale;
                    if (key0 + c > qrow) s = -1e30f;
                    rp[c] = s;
                    mx = fmaxf(mx, s);
                }
                float fac = __expf(mo - mx);
                float sum = St[wave][16 + lane] * fac;
#pragma unroll
                for (int c = 0; c < 32; ++c) {
                    float p = __expf(rp[c] - mx);
                    rp[c] = p;
                    sum += p;
                }
                St[wave][lane]      = mx;
                St[wave][16 + lane] = sum;
                St[wave][32 + lane] = fac;
            }
            __builtin_amdgcn_wave_barrier();

            // rescale running O
#pragma unroll
            for (int r = 0; r < 8; ++r) {
                float fac = St[wave][32 + r + hi * 8];
                oacc0[r] *= fac;
                oacc1[r] *= fac;
            }

            // P fragment (A-layout, f32 -> packed bf16)
            union { v16bf v; u32 d[8]; } pf;
            {
                const float* rp = Sw[wave] + l * SSTR;
#pragma unroll
                for (int p = 0; p < 8; ++p) {
                    int k = frag_k(p, hi);
                    pf.d[p] = f2bf_pk(rp[k], rp[k + 1]);
                }
            }
            // V fragments (B-layout) from global
            union { v16bf v; u16 u[16]; } vf0, vf1;
            {
                const u16* vp = vbase + (size_t)key0 * 384;
#pragma unroll
                for (int p = 0; p < 8; ++p) {
                    int k = frag_k(p, hi);
                    vf0.u[2 * p]     = vp[(size_t)k * 384 + l];
                    vf0.u[2 * p + 1] = vp[(size_t)(k + 1) * 384 + l];
                    vf1.u[2 * p]     = vp[(size_t)k * 384 + 16 + l];
                    vf1.u[2 * p + 1] = vp[(size_t)(k + 1) * 384 + 16 + l];
                }
            }
            oacc0 = wmma_bf16(pf.v, vf0.v, oacc0);
            oacc1 = wmma_bf16(pf.v, vf1.v, oacc1);
            __builtin_amdgcn_wave_barrier();
        }

        // normalize + store
#pragma unroll
        for (int r = 0; r < 8; ++r) {
            float inv = 1.0f / St[wave][16 + r + hi * 8];
            int grow = seq * SEQ_N + qrow0 + r + hi * 8;
            o[(size_t)grow * DMODEL + h * HDIM + l]      = f2bf(oacc0[r] * inv);
            o[(size_t)grow * DMODEL + h * HDIM + 16 + l] = f2bf(oacc1[r] * inv);
        }
    }
}

// ---------------------------------------------------------------------------
// Output heads: out = y @ out_w + out_b (5), reg = y @ reg_w + reg_b (3)
// ---------------------------------------------------------------------------
__global__ void k_head(const u16* __restrict__ y,
                       const float* __restrict__ ow, const float* __restrict__ ob,
                       const float* __restrict__ rw, const float* __restrict__ rb,
                       float* __restrict__ outbuf, float* __restrict__ regout) {
    int row = blockIdx.x * blockDim.x + threadIdx.x;
    if (row >= M_TOT) return;
    float a[5] = {0.f, 0.f, 0.f, 0.f, 0.f};
    float rg[3] = {0.f, 0.f, 0.f};
    const u16* yr = y + (size_t)row * DMODEL;
    for (int k = 0; k < DMODEL; ++k) {
        float v = bf2f(yr[k]);
#pragma unroll
        for (int c = 0; c < 5; ++c) a[c] += v * ow[k * 5 + c];
#pragma unroll
        for (int c = 0; c < 3; ++c) rg[c] += v * rw[k * 3 + c];
    }
#pragma unroll
    for (int c = 0; c < 5; ++c) outbuf[(size_t)row * 5 + c] = a[c] + ob[c];
#pragma unroll
    for (int c = 0; c < 3; ++c) regout[(size_t)row * 3 + c] = rg[c] + rb[c];
}

// ---------------------------------------------------------------------------
// Final assembly: eta_next / logits / qs_pred
//   d_out layout: [logits (M*6)] [qs_pred (M)] [regime (M*3)]
// ---------------------------------------------------------------------------
#define OFF_QS  (M_TOT * 6)

__global__ void k_final(const float* __restrict__ x, const float* __restrict__ outbuf,
                        float* __restrict__ dout) {
    int row = blockIdx.x * blockDim.x + threadIdx.x;
    if (row >= M_TOT) return;
    int n = row & (SEQ_N - 1);
    int baseR = row - n;                       // node 0 of this (b,t)
    const float* xr = x + (size_t)row * 6;
    float eta = xr[0], xn = xr[3], Qn = xr[4];
    const float* orow = outbuf + (size_t)row * 5;
    float qs = orow[0], inc = orow[1], hn = orow[2], un = orow[3], cov = orow[4];

    float dx = (x[(size_t)(baseR + 1) * 6 + 3] - x[(size_t)baseR * 6 + 3]) * 100.0f;
    float dq;
    if (n == 0) dq = (outbuf[(size_t)(baseR + 1) * 5] - qs) / dx;
    else        dq = (qs - outbuf[(size_t)(row - 1) * 5]) / dx;

    float eta_next = eta - 0.01f * 0.4f * dq - 0.01f * 0.002f * inc;

    float* lg = dout + (size_t)row * 6;
    lg[0] = eta_next; lg[1] = hn; lg[2] = un; lg[3] = xn; lg[4] = Qn; lg[5] = cov;
    dout[OFF_QS + row] = qs;
}

// ---------------------------------------------------------------------------
// Host launcher
// ---------------------------------------------------------------------------
extern "C" void kernel_launch(void* const* d_in, const int* in_sizes, int n_in,
                              void* d_out, int out_size, void* d_ws, size_t ws_size,
                              hipStream_t stream) {
    (void)in_sizes; (void)n_in; (void)out_size; (void)ws_size;

    const float* x    = (const float*)d_in[0];
    const float* npw  = (const float*)d_in[1];
    const float* npb  = (const float*)d_in[2];
    const float* temb = (const float*)d_in[3];
    const float* LP[NLAYER][16];
    for (int L = 0; L < NLAYER; ++L)
        for (int i = 0; i < 16; ++i)
            LP[L][i] = (const float*)d_in[4 + L * 16 + i];
    const float* lnf_g = (const float*)d_in[68];
    const float* lnf_b = (const float*)d_in[69];
    const float* out_w = (const float*)d_in[70];
    const float* out_b = (const float*)d_in[71];
    const float* reg_w = (const float*)d_in[72];
    const float* reg_b = (const float*)d_in[73];

    // workspace carve-up
    char* p = (char*)d_ws;
    auto alloc = [&](size_t bytes) -> char* {
        char* r = p;
        p += (bytes + 255) & ~(size_t)255;
        return r;
    };
    float* featf = (float*)alloc((size_t)M_TOT * DMODEL * 4);
    u16*   featb = (u16*)  alloc((size_t)M_TOT * DMODEL * 2);
    u16*   ybuf  = (u16*)  alloc((size_t)M_TOT * DMODEL * 2);
    u16*   mbuf  = (u16*)  alloc((size_t)M_TOT * DMODEL * 2);
    u16*   qkvb  = (u16*)  alloc((size_t)M_TOT * 384 * 2);
    u16*   obuf  = (u16*)  alloc((size_t)M_TOT * DMODEL * 2);
    u16*   hid   = (u16*)  alloc((size_t)M_TOT * FFDIM * 2);
    float* outb  = (float*)alloc((size_t)M_TOT * 5 * 4);

    // bf16 weight mirrors: per layer [ex_w1, ex_w2, qkv_w, proj_w, mlp_w1, mlp_w2]
    static const int widx[6] = {0, 2, 6, 8, 12, 14};
    static const int wsz[6]  = {257 * 128, 128 * 128, 128 * 384,
                                128 * 128, 128 * 2048, 2048 * 128};
    u16* WB[NLAYER][6];
    for (int L = 0; L < NLAYER; ++L)
        for (int i = 0; i < 6; ++i)
            WB[L][i] = (u16*)alloc((size_t)wsz[i] * 2);

    // 1) convert weights
    for (int L = 0; L < NLAYER; ++L)
        for (int i = 0; i < 6; ++i)
            k_cvt<<<(wsz[i] + 255) / 256, 256, 0, stream>>>(LP[L][widx[i]], WB[L][i], wsz[i]);

    // 2) embedding
    k_embed<<<(M_TOT * DMODEL + 255) / 256, 256, 0, stream>>>(x, npw, npb, temb, featf, featb);

    const dim3 gD(M_TOT / GM, DMODEL / GN);     // (192, 2)
    const dim3 gQKV(M_TOT / GM, 384 / GN);      // (192, 6)
    const dim3 gFF(M_TOT / GM, FFDIM / GN);     // (192, 32)

    // 3) transformer layers
    for (int L = 0; L < NLAYER; ++L) {
        const float* const* P = LP[L];
        // exchange: m = gelu([roll(feat),feat] @ ex_w1[0:256] + u*ex_w1[256] + b1)
        k_gemm<1, 1, true><<<gD, 256, 0, stream>>>(
            featb, DMODEL, WB[L][0], DMODEL, 256, P[1],
            nullptr, mbuf, DMODEL, x, P[0] + 256 * DMODEL);
        // feat += m @ ex_w2 + b2
        k_gemm<0, 2, false><<<gD, 256, 0, stream>>>(
            mbuf, DMODEL, WB[L][1], DMODEL, DMODEL, P[3],
            featf, featb, DMODEL, nullptr, nullptr);
        // ln1
        k_ln<<<M_TOT, 128, 0, stream>>>(featf, P[4], P[5], ybuf);
        // qkv
        k_gemm<0, 0, false><<<gQKV, 256, 0, stream>>>(
            ybuf, DMODEL, WB[L][2], 384, DMODEL, P[7],
            nullptr, qkvb, 384, nullptr, nullptr);
        // attention
        k_attn<<<dim3(M_TOT / SEQ_N, NHEAD), 256, 0, stream>>>(qkvb, obuf);
        // feat += o @ proj_w + proj_b
        k_gemm<0, 2, false><<<gD, 256, 0, stream>>>(
            obuf, DMODEL, WB[L][3], DMODEL, DMODEL, P[9],
            featf, featb, DMODEL, nullptr, nullptr);
        // ln2
        k_ln<<<M_TOT, 128, 0, stream>>>(featf, P[10], P[11], ybuf);
        // mlp
        k_gemm<0, 1, false><<<gFF, 256, 0, stream>>>(
            ybuf, DMODEL, WB[L][4], FFDIM, DMODEL, P[13],
            nullptr, hid, FFDIM, nullptr, nullptr);
        k_gemm<0, 2, false><<<gD, 256, 0, stream>>>(
            hid, FFDIM, WB[L][5], DMODEL, FFDIM, P[15],
            featf, featb, DMODEL, nullptr, nullptr);
    }

    // 4) final LN + heads + assembly
    k_ln<<<M_TOT, 128, 0, stream>>>(featf, lnf_g, lnf_b, ybuf);
    float* dout = (float*)d_out;
    k_head<<<(M_TOT + 255) / 256, 256, 0, stream>>>(
        ybuf, out_w, out_b, reg_w, reg_b, outb, dout + M_TOT * 6 + M_TOT);
    k_final<<<(M_TOT + 255) / 256, 256, 0, stream>>>(x, outb, dout);
}
